// TimeSpectralGradients_62062277427689
// MI455X (gfx1250) — matrix operand order repaired and checked by reading, not compiled, not run on hardware
//
#include <hip/hip_runtime.h>
#include <hip/hip_bf16.h>

typedef __attribute__((ext_vector_type(16))) _Float16 v16h;
typedef __attribute__((ext_vector_type(8)))  _Float16 v8h;
typedef __attribute__((ext_vector_type(8)))  float    v8f;

#define FS_F   100.0f
#define NSEG   64
#define NBAND  32      // frequency bands (scan length S, == WMMA K in k_sg)
#define BB     64      // batch
#define NN     3000    // time samples
#define NNP    3008    // padded: 94*32 and 188*16
#define MM     16      // classifier outputs (== WMMA M)
#define QF     5.0f

union HU { v8h h8[2]; v16h h16; };

// ---------------------------------------------------------------------------
// Kernel A: 31 lowpass biquads (band 31 is the masked zero baseline).
// One block per batch element b (x[b][n] is a uniform/broadcast load),
// one lane per band s -> E[b][n][s] stores are 128B fully coalesced.
// ---------------------------------------------------------------------------
__global__ void k_biquad(const float* __restrict__ x, float* __restrict__ E) {
  const int b = blockIdx.x;
  const int s = threadIdx.x;                 // 0..31
  const float* xb = x + (size_t)b * NN;
  float* eb = E + (size_t)b * NN * NBAND + s;

  if (s == NBAND - 1) {                      // step == 0 -> masked to zero
    for (int n = 0; n < NN; ++n) eb[(size_t)n * NBAND] = 0.0f;
    return;
  }
  const float cutoff = (float)(NBAND - 1 - s) * (FS_F / (float)NSEG);
  const float w0 = 2.0f * 3.14159265358979323846f * cutoff / FS_F;
  const float cw = cosf(w0), sw = sinf(w0);
  const float alpha = sw / (2.0f * QF);
  const float inv  = 1.0f / (1.0f + alpha);
  const float b1c  = (1.0f - cw) * inv;
  const float b0c  = 0.5f * b1c;             // b2 == b0
  const float a1c  = -2.0f * cw * inv;
  const float a2c  = (1.0f - alpha) * inv;

  float x1 = 0.f, x2 = 0.f, y1 = 0.f, y2 = 0.f;
  for (int n = 0; n < NN; ++n) {
    const float xt = xb[n];
    // critical path = one FMA on y1 (partial depends only on y2, 2 back)
    const float partial = fmaf(b0c, xt + x2, fmaf(b1c, x1, -a2c * y2));
    const float y = fmaf(-a1c, y1, partial);
    x2 = x1; x1 = xt; y2 = y1; y1 = y;
    eb[(size_t)n * NBAND] = y;
  }
}

// ---------------------------------------------------------------------------
// Kernel B: telescoping diffs (f32) -> f16 in two layouts, zero-padded to NNP.
//   Dh[b][n][s] : contiguous in s  (kernel D B-fragments, 32B vector loads)
//   Dt[b][s][n] : contiguous in n  (kernel C A-fragments, 16B vector loads)
// ---------------------------------------------------------------------------
__global__ void k_cvt(const float* __restrict__ x, const float* __restrict__ E,
                      _Float16* __restrict__ Dh, _Float16* __restrict__ Dt) {
  const int n = blockIdx.x * blockDim.x + threadIdx.x;
  const int b = blockIdx.y;
  if (n >= NNP) return;

  _Float16 d[NBAND];
  if (n < NN) {
    const float* row = E + ((size_t)b * NN + n) * NBAND;
    float prev = x[(size_t)b * NN + n];
#pragma unroll
    for (int s = 0; s < NBAND; ++s) {
      const float e = row[s];
      d[s] = (_Float16)(prev - e);
      prev = e;
    }
  } else {
#pragma unroll
    for (int s = 0; s < NBAND; ++s) d[s] = (_Float16)0.0f;
  }

  v8h* dst = (v8h*)(Dh + ((size_t)b * NNP + n) * NBAND);
#pragma unroll
  for (int q = 0; q < 4; ++q) {
    v8h v;
#pragma unroll
    for (int i = 0; i < 8; ++i) v[i] = d[q * 8 + i];
    dst[q] = v;
  }
  _Float16* tcol = Dt + (size_t)b * NBAND * NNP + n;
#pragma unroll
  for (int s = 0; s < NBAND; ++s) tcol[(size_t)s * NNP] = d[s];
}

// ---------------------------------------------------------------------------
// Kernel B2: WhT[m][n] = f16(Wf[n][m]), zero-padded to NNP.
// ---------------------------------------------------------------------------
__global__ void k_wfcvt(const float* __restrict__ Wf, _Float16* __restrict__ WhT) {
  const int n = blockIdx.x * blockDim.x + threadIdx.x;
  const int m = blockIdx.y;
  if (n >= NNP) return;
  WhT[(size_t)m * NNP + n] = (n < NN) ? (_Float16)Wf[n * MM + m] : (_Float16)0.0f;
}

// ---------------------------------------------------------------------------
// Kernel C: WbT[b][m][s] = | sum_n diff[b][s][n] * Wf[n][m] |  (f16 out)
// One wave per (s-half, b); 94 unconditional K=32 WMMA iterations.
// A layout (16-bit A 16x32, contiguous k as B): lane<16 row=lane K={0..15},
// lane>=16 row=lane-16 K={16..31}.  B: lane%16=col, K={0..15|16..31}.
// ---------------------------------------------------------------------------
__global__ void k_wband(const _Float16* __restrict__ Dt,
                        const _Float16* __restrict__ WhT,
                        _Float16* __restrict__ WbT) {
  const int shalf = blockIdx.x;              // 0..1
  const int b     = blockIdx.y;              // 0..63
  const int lane  = threadIdx.x;             // 0..31
  const int rlow  = lane & 15;
  const int hi    = lane >> 4;
  const int lo    = hi * 8;
  const int s     = shalf * 16 + rlow;       // A row (band)
  const int m     = rlow;                    // B col (output)

  const _Float16* arow = Dt + ((size_t)b * NBAND + s) * NNP;
  const _Float16* brow = WhT + (size_t)m * NNP + hi * 16;

  v8f c = {};
  for (int n0 = 0; n0 < NNP; n0 += 32) {     // 94 iterations, no guards
    HU ua;
    ua.h8[0] = *(const v8h*)(arow + n0 + hi * 16);        // K = hi*16+0..7
    ua.h8[1] = *(const v8h*)(arow + n0 + hi * 16 + 8);    // K = hi*16+8..15
    const v16h bm = *(const v16h*)(brow + n0);       // K = hi*16+0..15
    c = __builtin_amdgcn_wmma_f32_16x16x32_f16(false, ua.h16, false, bm,
                                               (short)0, c, false, false);
  }
  // C/D layout: VGPR r -> row r + 8*hi, col = lane%16
#pragma unroll
  for (int r = 0; r < 8; ++r) {
    const int srow = shalf * 16 + r + hi * 8;
    WbT[((size_t)b * MM + m) * NBAND + srow] = (_Float16)fabsf(c[r]);
  }
}

// ---------------------------------------------------------------------------
// Kernel D: SG[b][m][n] = Wf[n][m] * sum_s W[b][s][m] * diff[b][n][s]
// K = 32 bands exactly -> ONE wmma per (b, 16-wide n tile). All fragment
// loads unconditional (pad), only the output store is guarded.
// ---------------------------------------------------------------------------
__global__ void k_sg(const _Float16* __restrict__ Dh,
                     const _Float16* __restrict__ WbT,
                     const float* __restrict__ Wf, float* __restrict__ out) {
  const int tile = blockIdx.x;               // 0..187
  const int b    = blockIdx.y;
  const int lane = threadIdx.x;
  const int rlow = lane & 15;
  const int hi   = lane >> 4;
  const int lo   = hi * 8;
  const int n    = tile * 16 + rlow;         // B col / output col (< NNP)

  const _Float16* wr = WbT + ((size_t)b * MM + rlow) * NBAND;
  HU ua;
  ua.h8[0] = *(const v8h*)(wr + hi * 16);            // K(s) = hi*16+0..7
  ua.h8[1] = *(const v8h*)(wr + hi * 16 + 8);        // K(s) = hi*16+8..15
  const v16h bm = *(const v16h*)(Dh + ((size_t)b * NNP + n) * NBAND + hi * 16);

  v8f c = {};
  c = __builtin_amdgcn_wmma_f32_16x16x32_f16(false, ua.h16, false, bm,
                                             (short)0, c, false, false);
  if (n < NN) {
    const float* wfp = Wf + n * MM + hi * 8;
    const float4 w0 = *(const float4*)(wfp);
    const float4 w1 = *(const float4*)(wfp + 4);
    const float wv[8] = {w0.x, w0.y, w0.z, w0.w, w1.x, w1.y, w1.z, w1.w};
#pragma unroll
    for (int r = 0; r < 8; ++r)
      out[((size_t)b * MM + (r + hi * 8)) * NN + n] = wv[r] * c[r];
  }
}

// ---------------------------------------------------------------------------
extern "C" void kernel_launch(void* const* d_in, const int* in_sizes, int n_in,
                              void* d_out, int out_size, void* d_ws, size_t ws_size,
                              hipStream_t stream) {
  const float* x  = (const float*)d_in[0];   // [64, 3000]
  const float* Wf = (const float*)d_in[1];   // [3000, 16]
  float* out = (float*)d_out;                // [64, 16, 3000]

  char* ws = (char*)d_ws;                    // all offsets 64B-aligned
  const size_t EB  = (size_t)BB * NN  * NBAND * sizeof(float);     // 24,576,000
  const size_t DHB = (size_t)BB * NNP * NBAND * sizeof(_Float16);  // 12,320,768
  const size_t WHB = (size_t)MM * NNP * sizeof(_Float16);          //     96,256
  float*    E   = (float*)ws;
  _Float16* Dh  = (_Float16*)(ws + EB);
  _Float16* Dt  = (_Float16*)(ws + EB + DHB);
  _Float16* WhT = (_Float16*)(ws + EB + 2 * DHB);
  _Float16* WbT = (_Float16*)(ws + EB + 2 * DHB + WHB);

  k_biquad<<<dim3(BB), dim3(NBAND), 0, stream>>>(x, E);
  k_cvt   <<<dim3((NNP + 255) / 256, BB), dim3(256), 0, stream>>>(x, E, Dh, Dt);
  k_wfcvt <<<dim3((NNP + 255) / 256, MM), dim3(256), 0, stream>>>(Wf, WhT);
  k_wband <<<dim3(2, BB), dim3(32), 0, stream>>>(Dt, WhT, WbT);
  k_sg    <<<dim3(NNP / 16, BB), dim3(32), 0, stream>>>(Dh, WbT, Wf, out);
}
